// Attention_7138235646586
// MI455X (gfx1250) — matrix operand order repaired and checked by reading, not compiled, not run on hardware
//
#include <hip/hip_runtime.h>
#include <math.h>

#define H_   12
#define N_   2048
#define C_   768
#define DH_  64
#define BN_  4096      // B*N
#define C3_  2304      // 3*C

typedef __attribute__((ext_vector_type(16))) __bf16        v16bf;
typedef __attribute__((ext_vector_type(8)))  float         v8f;
typedef __attribute__((ext_vector_type(4))) unsigned int  u32x4;
typedef __attribute__((ext_vector_type(8)))  int           i32x8;
typedef __attribute__((ext_vector_type(4)))  int           i32x4;
typedef unsigned short u16;

// Padded LDS B-panel: 64 rows x 32 halves, +4 dwords pad every 16 dwords (TDM pad)
// -> row stride 40 halves (80 B): row r starts at bank (r*20)%64, all 16 distinct.
#define BROW_ 40
#define BBUF_ 2560          // 64 * 40 halves = 5120 B per buffer

__device__ __forceinline__ u16 f2bf(float f) {
  unsigned int u = __builtin_bit_cast(unsigned int, f);
  u += 0x7fffu + ((u >> 16) & 1u);          // round-to-nearest-even
  return (u16)(u >> 16);
}

union V16U { u32x4 q[2]; v16bf v; };

// A-matrix 16x32 bf16 tile from row-major [rows x ld]; rows row0..row0+15, K = k0..k0+31.
__device__ __forceinline__ v16bf loadA16x32(const u16* base, int ld, int row0, int k0) {
  int lane = threadIdx.x & 31;
  int r = lane & 15, sel = lane >> 4;
  const u16* p = base + (size_t)(row0 + r) * ld + k0 + sel * 8;
  V16U t;
  t.q[0] = *(const u32x4*)(p);
  t.q[1] = *(const u32x4*)(p + 16);
  return t.v;
}

// B-matrix 32x16 bf16 tile, sourced from B^T stored row-major [N x ld].
__device__ __forceinline__ v16bf loadB32x16(const u16* baseT, int ld, int n0, int k0) {
  int lane = threadIdx.x & 31;
  int r = lane & 15, sel = lane >> 4;
  const u16* p = baseT + (size_t)(n0 + r) * ld + k0 + sel * 8;
  V16U t;
  t.q[0] = *(const u32x4*)(p);
  t.q[1] = *(const u32x4*)(p + 16);
  return t.v;
}

__device__ __forceinline__ v8f wmma_bf16(v16bf a, v16bf b, v8f c) {
  return __builtin_amdgcn_wmma_f32_16x16x32_bf16(false, a, false, b, (short)0, c, false, false);
}

// ---- Tensor Data Mover: 2-D tile (32 x 64 bf16) of a row-major tensor -> LDS ----
// Descriptor per CDNA5 ISA. Groups 2/3 zero (<=2D tile);
// backend emits the 2-SGPR-group form: tensor_load_to_lds s[g0], s[g1].
__device__ __forceinline__ void tdm_load_B(unsigned lds_off, const u16* gsrc,
                                           unsigned td0, unsigned td1, unsigned stride0) {
  unsigned long long ga = (unsigned long long)(uintptr_t)(const void*)gsrc;
  u32x4 g0;
  g0[0] = 1u;                                             // count=1, user mode, no gather
  g0[1] = lds_off;                                        // lds_addr (bytes)
  g0[2] = (unsigned)ga;                                   // global_addr[31:0]
  g0[3] = (unsigned)((ga >> 32) & 0x01ffffffu) | 0x80000000u; // global_addr[56:32] | type=2
  i32x8 g1;
  // data_size=1 (2B), pad_enable, pad_interval=3 (16 dwords), pad_amount=3 (4 dwords)
  g1[0] = (int)((1u << 16) | (1u << 20) | (3u << 22) | (3u << 25));
  g1[1] = (int)((td0 & 0xffffu) << 16);                   // tensor_dim0[15:0]
  g1[2] = (int)((td0 >> 16) | ((td1 & 0xffffu) << 16));   // tensor_dim0[31:16] | tensor_dim1[15:0]
  g1[3] = (int)((td1 >> 16) | (32u << 16));               // tensor_dim1[31:16] | tile_dim0=32
  g1[4] = (int)64u;                                       // tile_dim1=64, tile_dim2=0
  g1[5] = (int)stride0;                                   // tensor_dim0_stride[31:0]
  g1[6] = 0;
  g1[7] = 0;
  i32x4 gz = {0, 0, 0, 0};
  i32x8 gz8 = {0, 0, 0, 0, 0, 0, 0, 0};
  __builtin_amdgcn_tensor_load_to_lds(g0, g1, gz, gz, gz8, 0);
}

// ---------------- conversion kernels ----------------
__global__ void k_cast_bf16(const float* __restrict__ in, u16* __restrict__ out, int n) {
  int i = (blockIdx.x * 256 + threadIdx.x) * 8;           // 8 elems/thread, b128 in/out
  if (i < n) {
    u32x4 f0 = *(const u32x4*)(in + i);
    u32x4 f1 = *(const u32x4*)(in + i + 4);
    union { u32x4 q; u16 s[8]; } pk;
    #pragma unroll
    for (int j = 0; j < 4; j++) {
      pk.s[j]     = f2bf(__builtin_bit_cast(float, f0[j]));
      pk.s[j + 4] = f2bf(__builtin_bit_cast(float, f1[j]));
    }
    *(u32x4*)(out + i) = pk.q;
  }
}

__global__ void k_transpose_bf16(const float* __restrict__ in, u16* __restrict__ out,
                                 int rows, int cols) {  // in[rows][cols] -> out[cols][rows]
  int i = blockIdx.x * 256 + threadIdx.x;
  if (i < rows * cols) {
    int r = i / cols, c = i - r * cols;
    out[(size_t)c * rows + r] = f2bf(in[i]);
  }
}

// ---------------- QKV GEMM: [4096x768] x [768x2304], TDM-staged B, scatter to q/k/vT ----------------
__global__ __launch_bounds__(256)
void k_qkv_gemm(const u16* __restrict__ xb, const u16* __restrict__ wT,
                u16* __restrict__ qb, u16* __restrict__ kb, u16* __restrict__ vT) {
  __shared__ u16 Bsh[2][BBUF_];
  int wave = threadIdx.x >> 5;
  int m0 = blockIdx.y * 128 + wave * 16;
  int n0 = blockIdx.x * 64;
  const v8f z = {0.f,0.f,0.f,0.f,0.f,0.f,0.f,0.f};
  v8f acc[4] = {z, z, z, z};

  unsigned lds0 = (unsigned)(uintptr_t)(void*)&Bsh[0][0];
  unsigned lds1 = (unsigned)(uintptr_t)(void*)&Bsh[1][0];
  const u16* bpanel = wT + (size_t)n0 * C_;

  if (wave == 0) tdm_load_B(lds0, bpanel, C_, C3_, C_);            // k0 = 0
  v16bf a_cur = loadA16x32(xb, C_, m0, 0);

  for (int k0 = 0; k0 < C_; k0 += 32) {
    int cur = (k0 >> 5) & 1;
    bool nxt = (k0 + 32) < C_;
    if (wave == 0) {
      if (nxt) {
        tdm_load_B(cur ? lds0 : lds1, bpanel + (k0 + 32), C_, C3_, C_);
        __builtin_amdgcn_s_wait_tensorcnt(1);                      // current buffer done
      } else {
        __builtin_amdgcn_s_wait_tensorcnt(0);
      }
    }
    __syncthreads();                                               // publish LDS tile
    v16bf a_next = a_cur;
    if (nxt) a_next = loadA16x32(xb, C_, m0, k0 + 32);             // prefetch next A tile
    const u16* bsrc = &Bsh[cur][0];
    v16bf bt[4];                                                   // batch LDS reads first:
    #pragma unroll
    for (int i = 0; i < 4; i++) bt[i] = loadB32x16(bsrc, BROW_, i * 16, 0);
    #pragma unroll
    for (int i = 0; i < 4; i++) acc[i] = wmma_bf16(a_cur, bt[i], acc[i]);
    a_cur = a_next;
    __syncthreads();                                               // buffer reuse guard
  }

  int lane = threadIdx.x & 31;
  int c16 = lane & 15, sel = lane >> 4;
  #pragma unroll
  for (int i = 0; i < 4; i++) {
    int col = n0 + i * 16 + c16;
    int three = col / C_;
    int rem = col - three * C_;
    int h = rem >> 6, dh = rem & 63;
    #pragma unroll
    for (int r = 0; r < 8; r++) {
      int row = m0 + r + 8 * sel;
      int b = row >> 11, n = row & 2047;
      size_t bh = (size_t)(b * H_ + h);
      float v = acc[i][r];
      if (three == 0)      qb[(bh * N_ + n) * DH_ + dh] = f2bf(v * 0.125f); // fold softmax scale
      else if (three == 1) kb[(bh * N_ + n) * DH_ + dh] = f2bf(v);
      else                 vT[(bh * DH_ + dh) * N_ + n] = f2bf(v);          // V stored transposed
    }
  }
}

// ---------------- flash attention: one 16-row q-tile per wave ----------------
__global__ __launch_bounds__(128)
void k_attention(const u16* __restrict__ qb, const u16* __restrict__ kb,
                 const u16* __restrict__ vT, const float* __restrict__ gate,
                 u16* __restrict__ attn) {
  __shared__ u16 Psh[4][16][32];           // per-wave P^T [qrow][key] bounce buffer
  int wv = threadIdx.x >> 5;
  int lane = threadIdx.x & 31;
  int qr = lane & 15, sel = lane >> 4;
  int h = blockIdx.y, b = blockIdx.z;
  int q0 = (blockIdx.x * 4 + wv) * 16;
  size_t bh = (size_t)(b * H_ + h);
  const u16* qp = qb + bh * (size_t)N_ * DH_;
  const u16* kp = kb + bh * (size_t)N_ * DH_;
  const u16* vp = vT + bh * (size_t)DH_ * N_;

  v16bf bq0 = loadB32x16(qp, DH_, q0, 0);
  v16bf bq1 = loadB32x16(qp, DH_, q0, 32);

  const v8f z = {0.f,0.f,0.f,0.f,0.f,0.f,0.f,0.f};
  v8f o[4] = {z, z, z, z};                 // O^T accum: reg r -> dh = i*16 + r + 8*sel
  float m = -INFINITY, lsum = 0.f;

  for (int j = 0; j < N_; j += 32) {       // 32 keys per step
    // S^T tiles = K . Q^T (contraction over dh, two K=32 WMMAs each)
    v8f s0 = z, s1 = z;
    s0 = wmma_bf16(loadA16x32(kp, DH_, j,       0), bq0, s0);
    s0 = wmma_bf16(loadA16x32(kp, DH_, j,      32), bq1, s0);
    s1 = wmma_bf16(loadA16x32(kp, DH_, j + 16,  0), bq0, s1);
    s1 = wmma_bf16(loadA16x32(kp, DH_, j + 16, 32), bq1, s1);

    // issue V^T tile loads early: global latency overlaps the exp math below
    v16bf av[4];
    #pragma unroll
    for (int i = 0; i < 4; i++) av[i] = loadA16x32(vp, N_, i * 16, j);

    // per-qrow (per-lane) online softmax; half-waves share a qrow
    float mloc = s0[0];
    #pragma unroll
    for (int r = 1; r < 8; r++) mloc = fmaxf(mloc, s0[r]);
    #pragma unroll
    for (int r = 0; r < 8; r++) mloc = fmaxf(mloc, s1[r]);
    mloc = fmaxf(mloc, __shfl_xor(mloc, 16, 32));
    float mnew  = fmaxf(m, mloc);
    float alpha = __expf(m - mnew);
    float ps = 0.f;
    asm volatile("" ::: "memory");         // order LDS stores after previous-iter LDS loads
    #pragma unroll
    for (int r = 0; r < 8; r++) {
      float p0 = __expf(s0[r] - mnew);
      float p1 = __expf(s1[r] - mnew);
      ps += p0 + p1;
      Psh[wv][qr][r + 8 * sel]      = f2bf(p0);   // key = j + r + 8*sel
      Psh[wv][qr][16 + r + 8 * sel] = f2bf(p1);   // key = j + 16 + r + 8*sel
    }
    ps += __shfl_xor(ps, 16, 32);
    lsum = lsum * alpha + ps;
    m = mnew;
    asm volatile("" ::: "memory");         // same-wave DS is in-order; stop compiler reordering
    v16bf bp = loadB32x16(&Psh[wv][0][0], 32, 0, 0);   // B^T = P^T row-major
    #pragma unroll
    for (int i = 0; i < 4; i++) {
      o[i] = o[i] * alpha;
      o[i] = wmma_bf16(av[i], bp, o[i]);   // O^T += V^T . P
    }
  }

  float scale = gate[h] / lsum;
  int n = q0 + qr;
  u16* orow = attn + ((size_t)b * N_ + n) * C_ + h * DH_;
  #pragma unroll
  for (int i = 0; i < 4; i++) {
    union { u32x4 q; u16 s[8]; } pk;
    #pragma unroll
    for (int r = 0; r < 8; r++) pk.s[r] = f2bf(o[i][r] * scale);
    *(u32x4*)(orow + i * 16 + 8 * sel) = pk.q;     // consecutive regs = consecutive dh
  }
}

// ---------------- projection GEMM: [4096x768] x [768x768] + bias -> f32, TDM-staged B ----------------
__global__ __launch_bounds__(256)
void k_proj_gemm(const u16* __restrict__ ab, const u16* __restrict__ wT,
                 const float* __restrict__ bias, float* __restrict__ out) {
  __shared__ u16 Bsh[2][BBUF_];
  int wave = threadIdx.x >> 5;
  int m0 = blockIdx.y * 128 + wave * 16;
  int n0 = blockIdx.x * 64;
  const v8f z = {0.f,0.f,0.f,0.f,0.f,0.f,0.f,0.f};
  v8f acc[4] = {z, z, z, z};

  unsigned lds0 = (unsigned)(uintptr_t)(void*)&Bsh[0][0];
  unsigned lds1 = (unsigned)(uintptr_t)(void*)&Bsh[1][0];
  const u16* bpanel = wT + (size_t)n0 * C_;

  if (wave == 0) tdm_load_B(lds0, bpanel, C_, C_, C_);
  v16bf a_cur = loadA16x32(ab, C_, m0, 0);

  for (int k0 = 0; k0 < C_; k0 += 32) {
    int cur = (k0 >> 5) & 1;
    bool nxt = (k0 + 32) < C_;
    if (wave == 0) {
      if (nxt) {
        tdm_load_B(cur ? lds0 : lds1, bpanel + (k0 + 32), C_, C_, C_);
        __builtin_amdgcn_s_wait_tensorcnt(1);
      } else {
        __builtin_amdgcn_s_wait_tensorcnt(0);
      }
    }
    __syncthreads();
    v16bf a_next = a_cur;
    if (nxt) a_next = loadA16x32(ab, C_, m0, k0 + 32);
    const u16* bsrc = &Bsh[cur][0];
    v16bf bt[4];
    #pragma unroll
    for (int i = 0; i < 4; i++) bt[i] = loadB32x16(bsrc, BROW_, i * 16, 0);
    #pragma unroll
    for (int i = 0; i < 4; i++) acc[i] = wmma_bf16(a_cur, bt[i], acc[i]);
    a_cur = a_next;
    __syncthreads();
  }

  int lane = threadIdx.x & 31;
  int c16 = lane & 15, sel = lane >> 4;
  #pragma unroll
  for (int i = 0; i < 4; i++) {
    int col = n0 + i * 16 + c16;
    float bv = bias[col];
    #pragma unroll
    for (int r = 0; r < 8; r++) {
      int row = m0 + r + 8 * sel;
      out[(size_t)row * C_ + col] = acc[i][r] + bv;
    }
  }
}

extern "C" void kernel_launch(void* const* d_in, const int* in_sizes, int n_in,
                              void* d_out, int out_size, void* d_ws, size_t ws_size,
                              hipStream_t stream) {
  const float* x      = (const float*)d_in[0];
  const float* w_qkv  = (const float*)d_in[1];
  const float* gate   = (const float*)d_in[2];
  const float* w_proj = (const float*)d_in[3];
  const float* b_proj = (const float*)d_in[4];

  char* ws = (char*)d_ws;                    // offsets in bytes (total ~36.2 MB)
  u16* xb     = (u16*)(ws + 0);              // 4096*768   bf16
  u16* wqkvT  = (u16*)(ws + 6291456);        // 2304*768   bf16
  u16* wprojT = (u16*)(ws + 9830400);        // 768*768    bf16
  u16* qb     = (u16*)(ws + 11010048);       // 24*2048*64 bf16
  u16* kb     = (u16*)(ws + 17301504);
  u16* vT     = (u16*)(ws + 23592960);
  u16* ab     = (u16*)(ws + 29884416);       // 4096*768   bf16

  k_cast_bf16<<<(BN_ * C_) / (256 * 8), 256, 0, stream>>>(x, xb, BN_ * C_);
  k_transpose_bf16<<<(C_ * C3_ + 255) / 256, 256, 0, stream>>>(w_qkv, wqkvT, C_, C3_);
  k_transpose_bf16<<<(C_ * C_ + 255) / 256, 256, 0, stream>>>(w_proj, wprojT, C_, C_);

  k_qkv_gemm<<<dim3(C3_ / 64, BN_ / 128), 256, 0, stream>>>(xb, wqkvT, qb, kb, vT);
  k_attention<<<dim3(N_ / 64, H_, 2), 128, 0, stream>>>(qb, kb, vT, gate, ab);
  k_proj_gemm<<<dim3(C_ / 64, BN_ / 128), 256, 0, stream>>>(ab, wprojT, b_proj, (float*)d_out);
}
